// GRUBaselineForecast_64037962383366
// MI455X (gfx1250) — matrix operand order repaired and checked by reading, not compiled, not run on hardware
//
#include <hip/hip_runtime.h>
#include <hip/hip_bf16.h>
#include <math.h>

// ---------------------------------------------------------------------------
// GRU forward, persistent-kernel version (2 waves/SIMD packing).
//   - One launch runs all T=1024 steps; device-wide atomic barrier per step.
//   - bf16 WMMA (v_wmma_f32_16x16x32_bf16), f32 accumulate.
//   - f32 hidden state lives in registers (each block owns a 32x64 tile).
//   - bf16 hidden state ping-pongs through global for cross-block K reads.
//   - 64 blocks x 256 threads: 8 waves/WGP = 2 waves/SIMD so one wave's WMMA
//     chain covers the other's post-fence L2-latency load stalls.
// ---------------------------------------------------------------------------

typedef __bf16  bf16_t;
typedef bf16_t  bfx8  __attribute__((ext_vector_type(8)));
typedef bf16_t  v16bf __attribute__((ext_vector_type(16)));
typedef float   v8f   __attribute__((ext_vector_type(8)));

#define NB   128   // batch
#define NT   1024  // time steps
#define ND   512   // input dim
#define NH   1024  // hidden dim
#define NBLOCKS 64 // (NH/64) * (NB/32)

// ---- WMMA fragment loaders (wave32, 16x16x32 bf16) ------------------------
// A (16x32, MxK): lane L holds row M=L&15; K runs [8*(L>>4)..+7] and +16.
__device__ __forceinline__ v16bf load_a_frag(const bf16_t* __restrict__ row,
                                             int half) {
  const bfx8 lo = *(const bfx8*)(row + 8 * half);
  const bfx8 hi = *(const bfx8*)(row + 16 + 8 * half);
  v16bf r;
#pragma unroll
  for (int i = 0; i < 8; ++i) { r[i] = lo[i]; r[i + 8] = hi[i]; }
  return r;
}

// B (32x16, KxN): lane L holds column N=L&15; K runs [8*(L>>4)..+7] and +16.
// `p` is already lane-adjusted to that starting element.
__device__ __forceinline__ v16bf load_b_frag(const bf16_t* __restrict__ p) {
  const bfx8 lo = *(const bfx8*)(p);
  const bfx8 hi = *(const bfx8*)(p + 16);
  v16bf r;
#pragma unroll
  for (int i = 0; i < 8; ++i) { r[i] = lo[i]; r[i + 8] = hi[i]; }
  return r;
}

__device__ __forceinline__ v8f wmma_bf16(v16bf a, v16bf b, v8f c) {
  return __builtin_amdgcn_wmma_f32_16x16x32_bf16(
      /*neg_a=*/false, a, /*neg_b=*/false, b,
      /*c_mod=*/(short)0, c, /*reuse_a=*/false, /*reuse_b=*/false);
}

// ---- f32 -> bf16 conversion -----------------------------------------------
__global__ void f32_to_bf16_kernel(const float* __restrict__ in,
                                   bf16_t* __restrict__ out, long n) {
  long i = (long)blockIdx.x * blockDim.x + threadIdx.x;
  long stride = (long)gridDim.x * blockDim.x;
  for (; i < n; i += stride) out[i] = (bf16_t)in[i];
}

// ---- zero hidden state + barrier words ------------------------------------
__global__ void zero_state_kernel(bf16_t* __restrict__ hb0,
                                  unsigned* __restrict__ bar, int n) {
  int i = blockIdx.x * blockDim.x + threadIdx.x;
  if (i < n) hb0[i] = (bf16_t)0.0f;
  if (i < 32) bar[i] = 0u;  // bar[0]=arrive count, bar[16]=generation
}

// ---- persistent GRU kernel ------------------------------------------------
// Grid: (NH/64, NB/32) = 64 blocks, 256 threads (8 waves).  Each wave owns
// one 16x16 output tile with 4 accumulator sets (r, z, n_hh, n_ih) plus the
// persistent f32 hidden-state fragment for that tile.
__global__ void __launch_bounds__(256)
gru_persistent_kernel(const bf16_t* __restrict__ xb,   // [NB, NT, ND] bf16
                      const bf16_t* __restrict__ wih,  // [3H, ND]     bf16
                      const bf16_t* __restrict__ whh,  // [3H, NH]     bf16
                      const float*  __restrict__ b,    // [3H]
                      const float*  __restrict__ bn,   // [H]
                      bf16_t*       __restrict__ hb0,  // [NB, NH] bf16
                      bf16_t*       __restrict__ hb1,  // [NB, NH] bf16
                      float*        __restrict__ h32,  // [NB, NH] f32 (final)
                      unsigned*     __restrict__ bar) {
  const int lane = threadIdx.x & 31;
  const int wave = threadIdx.x >> 5;
  const int wm   = wave >> 2;         // 0..1  row sub-tile
  const int wn   = wave & 3;          // 0..3  col sub-tile
  const int half = lane >> 4;         // 0..1
  const int nl   = lane & 15;

  const int m0 = blockIdx.y * 32 + wm * 16;        // wave row base
  const int j  = blockIdx.x * 64 + wn * 16 + nl;   // this lane's column

  // Loop-invariant pointers.
  const int bk = 8 * half;                         // lane K start for B frags
  const bf16_t* wr = whh + (size_t)(j)          * NH + bk;
  const bf16_t* wz = whh + (size_t)(NH + j)     * NH + bk;
  const bf16_t* wq = whh + (size_t)(2 * NH + j) * NH + bk;
  const bf16_t* cr = wih + (size_t)(j)          * ND + bk;
  const bf16_t* cz = wih + (size_t)(NH + j)     * ND + bk;
  const bf16_t* cq = wih + (size_t)(2 * NH + j) * ND + bk;
  const bf16_t* xa = xb + (size_t)(m0 + nl) * NT * ND;     // += ND per step
  const size_t  ha_off = (size_t)(m0 + nl) * NH;

  // Loop-invariant biases.
  const float bjr = b[j];
  const float bjz = b[NH + j];
  const float bjn = b[2 * NH + j];
  const float bnj = bn[j];

  // Persistent f32 hidden-state fragment (this wave owns rows m0..m0+15,
  // col j; no other block touches them).
  float hreg[8];
#pragma unroll
  for (int v = 0; v < 8; ++v) hreg[v] = 0.0f;

  unsigned gen_seen = 0u;

  for (int t = 0; t < NT; ++t) {
    const bf16_t* hb_old = (t & 1) ? hb1 : hb0;
    bf16_t*       hb_new = (t & 1) ? hb0 : hb1;
    const bf16_t* ha = hb_old + ha_off;

    // Prefetch next step's x row (contiguous, keeps L2 warm).
    __builtin_prefetch(xa + ND, 0, 3);

    v8f acc_r = {}, acc_z = {}, acc_nh = {}, acc_nx = {};

    // K segment 1: recurrent part (h @ w_hh.T), K = NH.
    for (int k = 0; k < NH; k += 32) {
      v16bf a   = load_a_frag(ha + k, half);
      v16bf brf = load_b_frag(wr + k);
      v16bf bzf = load_b_frag(wz + k);
      v16bf bqf = load_b_frag(wq + k);
      acc_r  = wmma_bf16(a, brf, acc_r);
      acc_z  = wmma_bf16(a, bzf, acc_z);
      acc_nh = wmma_bf16(a, bqf, acc_nh);
    }
    // K segment 2: input part (x_t @ w_ih.T), K = ND; n-gate kept separate.
    for (int k = 0; k < ND; k += 32) {
      v16bf a   = load_a_frag(xa + k, half);
      v16bf brf = load_b_frag(cr + k);
      v16bf bzf = load_b_frag(cz + k);
      v16bf bqf = load_b_frag(cq + k);
      acc_r  = wmma_bf16(a, brf, acc_r);
      acc_z  = wmma_bf16(a, bzf, acc_z);
      acc_nx = wmma_bf16(a, bqf, acc_nx);
    }

    // Gates + state update.  C/D layout: VGPR v, lanes0-15 -> M=m0+v,
    // lanes16-31 -> M=m0+v+8; N = lane&15 (== column j).
#pragma unroll
    for (int v = 0; v < 8; ++v) {
      const int m = m0 + v + 8 * half;
      const float r = 1.0f / (1.0f + __expf(-(acc_r[v] + bjr)));
      const float z = 1.0f / (1.0f + __expf(-(acc_z[v] + bjz)));
      const float n = tanhf((acc_nx[v] + bjn) + r * (acc_nh[v] + bnj));
      const float h_new = n + z * (hreg[v] - n);
      hreg[v] = h_new;
      hb_new[(size_t)m * NH + j] = (bf16_t)h_new;
    }

    // ---- device-wide barrier: all blocks must finish step t before t+1 ----
    __builtin_amdgcn_fence(__ATOMIC_RELEASE, "agent");  // flush this wave's stores
    __syncthreads();
    if (threadIdx.x == 0) {
      unsigned* cnt = bar;        // arrive counter
      unsigned* gen = bar + 16;   // generation (separate cacheline)
      const unsigned old = __hip_atomic_fetch_add(cnt, 1u, __ATOMIC_ACQ_REL,
                                                  __HIP_MEMORY_SCOPE_AGENT);
      if (old == (unsigned)(NBLOCKS - 1)) {
        __hip_atomic_store(cnt, 0u, __ATOMIC_RELAXED,
                           __HIP_MEMORY_SCOPE_AGENT);
        __hip_atomic_store(gen, gen_seen + 1u, __ATOMIC_RELEASE,
                           __HIP_MEMORY_SCOPE_AGENT);
      } else {
        while (__hip_atomic_load(gen, __ATOMIC_ACQUIRE,
                                 __HIP_MEMORY_SCOPE_AGENT) == gen_seen)
          __builtin_amdgcn_s_sleep(1);
      }
    }
    __syncthreads();
    __builtin_amdgcn_fence(__ATOMIC_ACQUIRE, "agent");  // inv L0/WGP$ for fresh h
    gen_seen += 1u;

    xa += ND;
  }

  // Spill final f32 hidden state for the output projection.
#pragma unroll
  for (int v = 0; v < 8; ++v) {
    const int m = m0 + v + 8 * half;
    h32[(size_t)m * NH + j] = hreg[v];
  }
}

// ---- final projection: out[b] = dot(h[b,:], w_out) + b_out ---------------
__global__ void __launch_bounds__(128)
gru_out_kernel(const float* __restrict__ h32, const float* __restrict__ w_out,
               const float* __restrict__ b_out, float* __restrict__ out) {
  __shared__ float red[128];
  const int row = blockIdx.x;
  float s = 0.0f;
  for (int k = threadIdx.x; k < NH; k += 128)
    s += h32[(size_t)row * NH + k] * w_out[k];
  red[threadIdx.x] = s;
  __syncthreads();
  for (int off = 64; off > 0; off >>= 1) {
    if (threadIdx.x < off) red[threadIdx.x] += red[threadIdx.x + off];
    __syncthreads();
  }
  if (threadIdx.x == 0) out[row] = red[0] + b_out[0];
}

// ---------------------------------------------------------------------------
extern "C" void kernel_launch(void* const* d_in, const int* in_sizes, int n_in,
                              void* d_out, int out_size, void* d_ws,
                              size_t ws_size, hipStream_t stream) {
  const float* x     = (const float*)d_in[0];  // [B, T, D]
  const float* w_ih  = (const float*)d_in[1];  // [3H, D]
  const float* w_hh  = (const float*)d_in[2];  // [3H, H]
  const float* b     = (const float*)d_in[3];  // [3H]
  const float* bn    = (const float*)d_in[4];  // [H]
  const float* w_out = (const float*)d_in[5];  // [1, H]
  const float* b_out = (const float*)d_in[6];  // [1]
  float* out = (float*)d_out;

  // Workspace partition (256B aligned regions), ~145 MB total.
  char* p = (char*)d_ws;
  auto carve = [&](size_t bytes) {
    char* r = p;
    p += (bytes + 255) & ~(size_t)255;
    return r;
  };
  bf16_t*   xb   = (bf16_t*)carve((size_t)NB * NT * ND * 2);
  bf16_t*   wihb = (bf16_t*)carve((size_t)3 * NH * ND * 2);
  bf16_t*   whhb = (bf16_t*)carve((size_t)3 * NH * NH * 2);
  float*    h32  = (float*) carve((size_t)NB * NH * 4);
  bf16_t*   hb0  = (bf16_t*)carve((size_t)NB * NH * 2);
  bf16_t*   hb1  = (bf16_t*)carve((size_t)NB * NH * 2);
  unsigned* bar  = (unsigned*)carve(32 * sizeof(unsigned));
  (void)ws_size; (void)in_sizes; (void)n_in; (void)out_size;

  // One-pass bf16 conversions (parallel, HBM-bandwidth bound).
  f32_to_bf16_kernel<<<4096, 256, 0, stream>>>(x, xb, (long)NB * NT * ND);
  f32_to_bf16_kernel<<<512, 256, 0, stream>>>(w_ih, wihb, (long)3 * NH * ND);
  f32_to_bf16_kernel<<<512, 256, 0, stream>>>(w_hh, whhb, (long)3 * NH * NH);
  zero_state_kernel<<<(NB * NH + 255) / 256, 256, 0, stream>>>(hb0, bar,
                                                               NB * NH);

  // Single persistent launch runs the whole recurrence.
  dim3 grid(NH / 64, NB / 32);  // 16 x 4 = 64 blocks
  gru_persistent_kernel<<<grid, 256, 0, stream>>>(xb, wihb, whhb, b, bn, hb0,
                                                  hb1, h32, bar);

  gru_out_kernel<<<NB, 128, 0, stream>>>(h32, w_out, b_out, out);
}
